// MultiHeadAttention_1666447311069
// MI455X (gfx1250) — matrix operand order repaired and checked by reading, not compiled, not run on hardware
//
#include <hip/hip_runtime.h>
#include <hip/hip_bf16.h>

// ---------------------------------------------------------------------------
// Types for CDNA5 WMMA (wave32): v_wmma_f32_16x16x32_bf16
// ---------------------------------------------------------------------------
typedef __bf16 bf16;
typedef __attribute__((ext_vector_type(16))) bf16  v16bf;
typedef __attribute__((ext_vector_type(8)))  bf16  v8bf;
typedef __attribute__((ext_vector_type(4)))  bf16  v4bf;
typedef __attribute__((ext_vector_type(8)))  float v8f;
typedef __attribute__((ext_vector_type(4)))  float v4f;

#define WMMA_BF16(a, b, c) \
  __builtin_amdgcn_wmma_f32_16x16x32_bf16(false, (a), false, (b), (short)0, (c), false, false)

__device__ __forceinline__ v16bf cat16(v8bf lo, v8bf hi) {
  return __builtin_shufflevector(lo, hi, 0,1,2,3,4,5,6,7,8,9,10,11,12,13,14,15);
}

__device__ __forceinline__ v16bf zero16bf() {
  v16bf z;
#pragma unroll
  for (int i = 0; i < 16; ++i) z[i] = (bf16)0.0f;
  return z;
}

__device__ __forceinline__ v4bf cvt4(v4f t) {
  v4bf x;
#pragma unroll
  for (int j = 0; j < 4; ++j) x[j] = (bf16)t[j];
  return x;
}

// A-fragment (16x32 bf16, M x K) from LDS, row-major rows of `stride` halves.
// lanes0-15 row=lane, K contiguous: element s -> K = 16*g + s.
__device__ __forceinline__ v16bf lds_a_frag(const bf16* base, int stride,
                                            int m0, int koff, int lane) {
  const int g = lane >> 4;
  const bf16* r = base + (m0 + (lane & 15)) * stride + koff;
  v8bf lo = *(const v8bf*)(r + 16 * g);
  v8bf hi = *(const v8bf*)(r + 16 * g + 8);
  return cat16(lo, hi);
}

// B-fragment (32x16 bf16, K x N) from LDS staged K-contiguous per column:
// BsT[n][k], lane column n = lane&15, halves h -> K = 16*g + h (contiguous 16).
__device__ __forceinline__ v16bf lds_bT_frag(const bf16* base, int stride,
                                             int n0, int koff, int lane) {
  const int g = lane >> 4;
  const bf16* r = base + (n0 + (lane & 15)) * stride + koff + 16 * g;
  v8bf lo = *(const v8bf*)(r);
  v8bf hi = *(const v8bf*)(r + 8);
  return cat16(lo, hi);
}

// B-fragment from 16 consecutive fp32 in global memory: 4x b128 loads, then cvt.
__device__ __forceinline__ v16bf glb_b_frag_f32(const float* __restrict__ p) {
  v4f t0 = *(const v4f*)(p + 0);
  v4f t1 = *(const v4f*)(p + 4);
  v4f t2 = *(const v4f*)(p + 8);
  v4f t3 = *(const v4f*)(p + 12);
  v16bf r;
#pragma unroll
  for (int j = 0; j < 4; ++j) {
    r[j + 0]  = (bf16)t0[j];
    r[j + 4]  = (bf16)t1[j];
    r[j + 8]  = (bf16)t2[j];
    r[j + 12] = (bf16)t3[j];
  }
  return r;
}

// ---------------------------------------------------------------------------
// GEMM: C[M,N] = alpha * (A[M,K] @ W[K,N] + bias[N]), fp32 in/out, bf16 WMMA.
// Block 256 = 8 waves; 128x64 tile; wave owns 32x32 (4 accumulators).
// Double-buffered LDS, one barrier per 32-deep K-step, b128 staging loads
// batched before conversion, L2 prefetch two tiles ahead.
// ---------------------------------------------------------------------------
#define AS_LD 48  // halves; 96B rows keep 16B alignment, break bank conflicts
#define GBM 128
#define GBN 64

__global__ __launch_bounds__(256) void gemm_bias_wmma(
    const float* __restrict__ A, const float* __restrict__ W,
    const float* __restrict__ bias, float* __restrict__ C,
    int M, int N, int K, float alpha) {
  __shared__ __align__(16) bf16 As[2][GBM * AS_LD];   // 2 x 12 KB
  __shared__ __align__(16) bf16 BsT[2][GBN * AS_LD];  // 2 x  6 KB

  const int tid = threadIdx.x;
  const int lane = tid & 31, w = tid >> 5;
  const int g = lane >> 4, ln = lane & 15;
  const int m0 = blockIdx.y * GBM, n0 = blockIdx.x * GBN;
  const int rp = w >> 1;  // rows rp*32 .. rp*32+31
  const int cp = w & 1;   // cols cp*32 .. cp*32+31

  v8f acc00 = {}, acc01 = {}, acc10 = {}, acc11 = {};

  // Batched b128 loads first (issue as a clause), then convert + LDS stores.
  auto stage = [&](int buf, int k0) {
    v4f ta[4], tb[2];
#pragma unroll
    for (int i = 0; i < 4; ++i) {            // A tile 128x32 = 1024 float4
      int e4 = tid + 256 * i;
      int m = e4 >> 3, kq = (e4 & 7) * 4;
      ta[i] = *(const v4f*)&A[(size_t)(m0 + m) * K + k0 + kq];
    }
#pragma unroll
    for (int i = 0; i < 2; ++i) {            // W tile 32x64 = 512 float4
      int e4 = tid + 256 * i;
      int kk = e4 >> 4, n4 = (e4 & 15) * 4;
      tb[i] = *(const v4f*)&W[(size_t)(k0 + kk) * N + n0 + n4];
    }
#pragma unroll
    for (int i = 0; i < 4; ++i) {
      int e4 = tid + 256 * i;
      int m = e4 >> 3, kq = (e4 & 7) * 4;
      *(v4bf*)&As[buf][m * AS_LD + kq] = cvt4(ta[i]);   // ds_store_b64
    }
#pragma unroll
    for (int i = 0; i < 2; ++i) {
      int e4 = tid + 256 * i;
      int kk = e4 >> 4, n4 = (e4 & 15) * 4;
#pragma unroll
      for (int j = 0; j < 4; ++j)                       // transposed scatter
        BsT[buf][(n4 + j) * AS_LD + kk] = (bf16)tb[i][j];
    }
  };

  stage(0, 0);
  int buf = 0;
  for (int k0 = 0; k0 < K; k0 += 32) {
    __syncthreads();
    if (k0 + 32 < K) stage(buf ^ 1, k0 + 32);
    if (k0 + 64 < K) {  // L2 prefetch of the tile after next (global_prefetch)
      int ea = tid * 16, ma = ea >> 5, ka = ea & 31;
      __builtin_prefetch(&A[(size_t)(m0 + ma) * K + k0 + 64 + ka], 0, 3);
      int eb = tid * 8, kb = eb >> 6, nb = eb & 63;
      __builtin_prefetch(&W[(size_t)(k0 + 64 + kb) * N + n0 + nb], 0, 3);
    }
    v16bf a0 = lds_a_frag(As[buf], AS_LD, rp * 32,      0, lane);
    v16bf a1 = lds_a_frag(As[buf], AS_LD, rp * 32 + 16, 0, lane);
    v16bf b0 = lds_bT_frag(BsT[buf], AS_LD, cp * 32,      0, lane);
    v16bf b1 = lds_bT_frag(BsT[buf], AS_LD, cp * 32 + 16, 0, lane);
    acc00 = WMMA_BF16(a0, b0, acc00);
    acc01 = WMMA_BF16(a0, b1, acc01);
    acc10 = WMMA_BF16(a1, b0, acc10);
    acc11 = WMMA_BF16(a1, b1, acc11);
    buf ^= 1;
  }

#pragma unroll
  for (int tr = 0; tr < 2; ++tr) {
#pragma unroll
    for (int tc = 0; tc < 2; ++tc) {
      v8f acc = tr ? (tc ? acc11 : acc10) : (tc ? acc01 : acc00);
      int n = n0 + cp * 32 + tc * 16 + ln;
      float bv = bias[n];
#pragma unroll
      for (int i = 0; i < 8; ++i) {
        int m = m0 + rp * 32 + tr * 16 + i + 8 * g;
        C[(size_t)m * N + n] = alpha * (acc[i] + bv);
      }
    }
  }
}

// ---------------------------------------------------------------------------
// Attention kernel: one workgroup = one (b, h, 32-query tile).
// LDS (242 KB of the 320 KB WGP pool):
//   q tile (bf16) | s_rel table (f32, WMMA-computed, gathered in-LDS) |
//   full 32x1024 score block (f32) | 32x1024 probs (bf16) |
//   double-buffered transposed V chunk (bf16).
// ---------------------------------------------------------------------------
#define S_LEN 1024
#define HD    64
#define HIDM  1024
#define RNUM  257
#define RPAD  272
#define QT    32
#define VS_LD 48

#define OFF_SREL (QT * HD * 2)
#define OFF_SC   (OFF_SREL + QT * RPAD * 4)
#define OFF_PR   (OFF_SC + QT * S_LEN * 4)
#define OFF_VS   (OFF_PR + QT * S_LEN * 2)
#define SMEM_TOT (OFF_VS + 2 * HD * VS_LD * 2)

__global__ __launch_bounds__(256) void attn_relpos_wmma(
    const float* __restrict__ qb, const float* __restrict__ kb,
    const float* __restrict__ vb, const unsigned char* __restrict__ mask,
    const int* __restrict__ rdist, const float* __restrict__ rel_emb,
    float* __restrict__ xb) {
  extern __shared__ __align__(16) char smem[];
  bf16*  q_s  = (bf16*)smem;
  float* srel = (float*)(smem + OFF_SREL);
  float* sc   = (float*)(smem + OFF_SC);
  bf16*  pr   = (bf16*)(smem + OFF_PR);
  bf16*  vs   = (bf16*)(smem + OFF_VS);

  const int tid = threadIdx.x;
  const int lane = tid & 31, w = tid >> 5;
  const int g = lane >> 4, ln = lane & 15;
  const int h = blockIdx.y, b = blockIdx.z;
  const int q0 = blockIdx.x * QT;

  // P0: stage 32x64 q tile as bf16 via b128 loads (q already carries 1/sqrt(d)).
  {
    v4f tq[2];
#pragma unroll
    for (int i = 0; i < 2; ++i) {           // 512 float4
      int e4 = tid + 256 * i;
      int m = e4 >> 4, d4 = (e4 & 15) * 4;
      tq[i] = *(const v4f*)&qb[(size_t)(b * S_LEN + q0 + m) * HIDM + h * HD + d4];
    }
#pragma unroll
    for (int i = 0; i < 2; ++i) {
      int e4 = tid + 256 * i;
      int m = e4 >> 4, d4 = (e4 & 15) * 4;
      *(v4bf*)&q_s[m * HD + d4] = cvt4(tq[i]);
    }
  }
  __syncthreads();

  // P1: s_rel[32 x 272] = q_tile(32x64) @ rel_emb^T(64x272), via WMMA.
  for (int t = w; t < 34; t += 8) {
    int rt = t & 1, ct = t >> 1;
    int r = ct * 16 + ln;
    v8f acc = {};
#pragma unroll
    for (int s = 0; s < 2; ++s) {
      v16bf a = lds_a_frag(q_s, HD, rt * 16, s * 32, lane);
      v16bf bb = (r < RNUM)
                   ? glb_b_frag_f32(rel_emb + (size_t)r * HD + s * 32 + 16 * g)
                   : zero16bf();
      acc = WMMA_BF16(a, bb, acc);
    }
#pragma unroll
    for (int i = 0; i < 8; ++i)
      srel[(rt * 16 + i + 8 * g) * RPAD + r] = acc[i];
  }
  __syncthreads();

  // P2: scores. Each wave owns 128 consecutive keys (8 chunks of 16).
  v16bf a00 = lds_a_frag(q_s, HD,  0,  0, lane);
  v16bf a01 = lds_a_frag(q_s, HD,  0, 32, lane);
  v16bf a10 = lds_a_frag(q_s, HD, 16,  0, lane);
  v16bf a11 = lds_a_frag(q_s, HD, 16, 32, lane);
  for (int c = w * 8; c < w * 8 + 8; ++c) {
    const int j0 = c * 16;
    const float* kp = kb + (size_t)(b * S_LEN + j0 + ln) * HIDM + h * HD + 16 * g;
    v16bf b0 = glb_b_frag_f32(kp);       // d = 0..31
    v16bf b1 = glb_b_frag_f32(kp + 32);  // d = 32..63
#pragma unroll
    for (int rt = 0; rt < 2; ++rt) {
      v8f acc = {};
      acc = WMMA_BF16(rt ? a10 : a00, b0, acc);
      acc = WMMA_BF16(rt ? a11 : a01, b1, acc);
      const int j = j0 + ln;
#pragma unroll
      for (int i = 0; i < 8; ++i) {
        int m = rt * 16 + i + 8 * g;
        size_t qg = (size_t)(b * S_LEN + q0 + m) * S_LEN + j;
        int dd = rdist[qg] + 128;                 // [0, 256]
        float sv = acc[i] + srel[m * RPAD + dd];  // rel-bias gather from LDS
        if (mask[qg]) sv = -__builtin_inff();
        sc[m * S_LEN + j] = sv;
      }
    }
  }
  __syncthreads();

  // P3: softmax, one wave per 4 rows; b128 row sweeps, packed b64 prob stores.
  for (int r = 0; r < 4; ++r) {
    const int m = w * 4 + r;
    const float* row = sc + m * S_LEN;
    float mx = -3.4e38f;
    for (int j4 = lane * 4; j4 < S_LEN; j4 += 128) {
      v4f t = *(const v4f*)(row + j4);
      mx = fmaxf(mx, fmaxf(fmaxf(t[0], t[1]), fmaxf(t[2], t[3])));
    }
#pragma unroll
    for (int off = 16; off > 0; off >>= 1) mx = fmaxf(mx, __shfl_xor(mx, off, 32));
    float sum = 0.0f;
    for (int j4 = lane * 4; j4 < S_LEN; j4 += 128) {
      v4f t = *(const v4f*)(row + j4);
      sum += __expf(t[0] - mx) + __expf(t[1] - mx) +
             __expf(t[2] - mx) + __expf(t[3] - mx);
    }
#pragma unroll
    for (int off = 16; off > 0; off >>= 1) sum += __shfl_xor(sum, off, 32);
    const float inv = 1.0f / sum;
    bf16* prow = pr + m * S_LEN;
    for (int j4 = lane * 4; j4 < S_LEN; j4 += 128) {
      v4f t = *(const v4f*)(row + j4);
      v4bf p4;
#pragma unroll
      for (int j = 0; j < 4; ++j) p4[j] = (bf16)(__expf(t[j] - mx) * inv);
      *(v4bf*)(prow + j4) = p4;
    }
  }

  // P4: out(32x64) = P(32x1024) @ V(1024x64). V chunks staged coalesced (b128)
  // into transposed, double-buffered LDS; wave -> one 16x16 tile.
  auto stage_v = [&](int buf, int kk) {
    bf16* dst = vs + buf * HD * VS_LD;
    v4f tv[2];
#pragma unroll
    for (int i = 0; i < 2; ++i) {           // 32 keys x 64 cols = 512 float4
      int e4 = tid + 256 * i;
      int kyl = e4 >> 4, col4 = (e4 & 15) * 4;
      tv[i] = *(const v4f*)&vb[(size_t)(b * S_LEN + kk + kyl) * HIDM + h * HD + col4];
    }
#pragma unroll
    for (int i = 0; i < 2; ++i) {
      int e4 = tid + 256 * i;
      int kyl = e4 >> 4, col4 = (e4 & 15) * 4;
#pragma unroll
      for (int j = 0; j < 4; ++j)
        dst[(col4 + j) * VS_LD + kyl] = (bf16)tv[i][j];
    }
  };

  {
    const int rt = w >> 2, ct = w & 3;
    v8f acc = {};
    stage_v(0, 0);
    int buf = 0;
    for (int kk = 0; kk < S_LEN; kk += 32) {
      __syncthreads();
      if (kk + 32 < S_LEN) stage_v(buf ^ 1, kk + 32);
      v16bf a  = lds_a_frag(pr, S_LEN, rt * 16, kk, lane);
      v16bf bb = lds_bT_frag(vs + buf * HD * VS_LD, VS_LD, ct * 16, 0, lane);
      acc = WMMA_BF16(a, bb, acc);
      buf ^= 1;
    }
    const int n = h * HD + ct * 16 + ln;
#pragma unroll
    for (int i = 0; i < 8; ++i) {
      int m = rt * 16 + i + 8 * g;
      xb[(size_t)(b * S_LEN + q0 + m) * HIDM + n] = acc[i];
    }
  }
}

// ---------------------------------------------------------------------------
// Launch: Q/K/V projections -> attention -> output projection.
// ---------------------------------------------------------------------------
extern "C" void kernel_launch(void* const* d_in, const int* in_sizes, int n_in,
                              void* d_out, int out_size, void* d_ws, size_t ws_size,
                              hipStream_t stream) {
  (void)in_sizes; (void)n_in; (void)out_size; (void)ws_size;

  const float*         query = (const float*)d_in[0];
  const float*         key_  = (const float*)d_in[1];
  const float*         value = (const float*)d_in[2];
  const unsigned char* amask = (const unsigned char*)d_in[3];
  const int*           rdist = (const int*)d_in[4];
  const float* Wq = (const float*)d_in[5];  const float* bq = (const float*)d_in[6];
  const float* Wk = (const float*)d_in[7];  const float* bk = (const float*)d_in[8];
  const float* Wv = (const float*)d_in[9];  const float* bv = (const float*)d_in[10];
  const float* Wo = (const float*)d_in[11]; const float* bo = (const float*)d_in[12];
  const float* rel = (const float*)d_in[13];
  float* out = (float*)d_out;

  const size_t SZ = (size_t)2 * 1024 * 1024;  // B*S*HID elements
  float* qb = (float*)d_ws;
  float* kb = qb + SZ;
  float* vb = kb + SZ;
  float* xb = vb + SZ;

  dim3 blk(256);
  dim3 gg(1024 / GBN, 2048 / GBM);  // 16 x 16 = 256 workgroups

  // 1/sqrt(HEAD_DIM) folded into q: scales both QK^T and s_rel, as in reference.
  gemm_bias_wmma<<<gg, blk, 0, stream>>>(query, Wq, bq, qb, 2048, 1024, 1024, 0.125f);
  gemm_bias_wmma<<<gg, blk, 0, stream>>>(key_,  Wk, bk, kb, 2048, 1024, 1024, 1.0f);
  gemm_bias_wmma<<<gg, blk, 0, stream>>>(value, Wv, bv, vb, 2048, 1024, 1024, 1.0f);

  (void)hipFuncSetAttribute((const void*)attn_relpos_wmma,
                            hipFuncAttributeMaxDynamicSharedMemorySize,
                            (int)SMEM_TOT);
  attn_relpos_wmma<<<dim3(1024 / QT, 16, 2), blk, SMEM_TOT, stream>>>(
      qb, kb, vb, amask, rdist, rel, xb);

  gemm_bias_wmma<<<gg, blk, 0, stream>>>(xb, Wo, bo, out, 2048, 1024, 1024, 1.0f);
}
